// RelativeSelfAttention_15418932593421
// MI455X (gfx1250) — matrix operand order repaired and checked by reading, not compiled, not run on hardware
//
#include <hip/hip_runtime.h>

// ---------------------------------------------------------------------------
// Fused Swin window attention for gfx1250 (MI455X), wave32 + WMMA bf16.
// One workgroup (256 threads = 8 waves) processes one 49-token window
// entirely in LDS: QKV GEMM -> per-head attention -> gamma GEMM -> proj GEMM.
// Weights are pre-converted to bf16 in d_ws (L2-resident across all blocks).
// ---------------------------------------------------------------------------

typedef __bf16 bf16;
typedef __attribute__((ext_vector_type(4)))  __bf16 v4bf;
typedef __attribute__((ext_vector_type(8)))  __bf16 v8bf;
typedef __attribute__((ext_vector_type(16))) __bf16 v16bf;
typedef __attribute__((ext_vector_type(8)))  float  v8f;

#define WIN_N   49
#define NPAD    64
#define CDIM    512
#define HEADS   16
#define HD      32
#define NWIN    2048

// LDS strides in elements (padded to break 64-bank conflicts, 16B-aligned rows)
#define XS_LD   520     // x / attn-out reuse region: 64 x 520 bf16
#define QK_LD   1032    // q|k rows: 64 x 1032 bf16 ; also reused for y (stride XS_LD)
#define VT_LD   72      // v^T: 512 x 72 bf16
#define SS_LD   68      // scores: 64 x 68 f32
#define PS_LD   72      // probs: 64 x 72 bf16

#define XS_OFF  0
#define QK_OFF  (XS_OFF + NPAD * XS_LD * 2)        // 66560
#define VT_OFF  (QK_OFF + NPAD * QK_LD * 2)        // 198656
#define SS_OFF  (VT_OFF + CDIM * VT_LD * 2)        // 272384
#define PS_OFF  (SS_OFF + NPAD * SS_LD * 4)        // 289792
#define SMEM_BYTES (PS_OFF + NPAD * PS_LD * 2)     // 299008  (< 320 KB WGP LDS)

// A-fragment (16x32 bf16): lane holds row li, K = half*16 + {0..15}
// (contiguous).  p = row_base + kstep*32 + half*16.
__device__ __forceinline__ v16bf frag_a(const bf16* p) {
  v8bf lo = *(const v8bf*)p;
  v8bf hi = *(const v8bf*)(p + 8);
  return __builtin_shufflevector(lo, hi, 0,1,2,3,4,5,6,7,8,9,10,11,12,13,14,15);
}
// B-fragment (32x16 bf16): lane holds column li, K = half*16 + {0..15}
// (contiguous).  p = colrow_base + kstep*32 + half*16.
__device__ __forceinline__ v16bf frag_b(const bf16* p) {
  v8bf lo = *(const v8bf*)p;
  v8bf hi = *(const v8bf*)(p + 8);
  return __builtin_shufflevector(lo, hi, 0,1,2,3,4,5,6,7,8,9,10,11,12,13,14,15);
}

__device__ __forceinline__ v8f wmma_bf16(v16bf a, v16bf b, v8f c) {
  return __builtin_amdgcn_wmma_f32_16x16x32_bf16(false, a, false, b, (short)0, c,
                                                 false, false);
}

// C = A(64x512, LDS row-major, stride lda) x B^T (B rows in global bf16, ld=512)
// for one 16-wide output column tile; acc[m] covers rows m*16..m*16+15.
// Unroll capped at 4 to keep the wave under 256 VGPRs (no VGPR-MSB switching).
__device__ __forceinline__ void gemm_nt64(const bf16* __restrict__ As, int lda,
                                          const bf16* __restrict__ Bg,
                                          int li, int half, v8f acc[4]) {
  const bf16* brow = Bg + li * CDIM + half * 16;
#pragma unroll 4
  for (int ks = 0; ks < 16; ++ks) {
    v16bf bfrag = frag_b(brow + ks * 32);
#pragma unroll
    for (int m = 0; m < 4; ++m) {
      const bf16* ap = As + (m * 16 + li) * lda + ks * 32 + half * 16;
      acc[m] = wmma_bf16(frag_a(ap), bfrag, acc[m]);
    }
  }
}

// ---------------------------------------------------------------------------
// Prep: convert weights fp32->bf16, gather padded relative-position bias.
// ---------------------------------------------------------------------------
extern "C" __global__ void prep_kernel(const float* __restrict__ wqkv,
                                       const float* __restrict__ wgam,
                                       const float* __restrict__ wprj,
                                       const float* __restrict__ btab,
                                       const int*   __restrict__ ridx,
                                       bf16* __restrict__ wqkv_b,
                                       bf16* __restrict__ wgam_b,
                                       bf16* __restrict__ wprj_b,
                                       float* __restrict__ biasB) {
  int tid = blockIdx.x * blockDim.x + threadIdx.x;
  int stride = gridDim.x * blockDim.x;
  for (int i = tid; i < 3 * CDIM * CDIM; i += stride) wqkv_b[i] = (bf16)wqkv[i];
  for (int i = tid; i < CDIM * CDIM; i += stride)     wgam_b[i] = (bf16)wgam[i];
  for (int i = tid; i < CDIM * CDIM; i += stride)     wprj_b[i] = (bf16)wprj[i];
  for (int i = tid; i < HEADS * NPAD * NPAD; i += stride) {
    int h = i >> 12, r = (i >> 6) & 63, c = i & 63;
    float v = 0.0f;
    if (r < WIN_N && c < WIN_N) v = btab[ridx[r * WIN_N + c] * HEADS + h];
    biasB[i] = v;
  }
}

// ---------------------------------------------------------------------------
// Fused per-window kernel.  __launch_bounds__(256, 4): min 4 waves/EU caps
// VGPRs at 256/wave so no s_set_vgpr_msb traffic is generated.
// ---------------------------------------------------------------------------
extern "C" __global__ void __launch_bounds__(256, 4)
win_attn_fused(const float* __restrict__ x,
               const float* __restrict__ b_qkv,
               const float* __restrict__ b_gam,
               const float* __restrict__ b_prj,
               const bf16*  __restrict__ wqkv,
               const bf16*  __restrict__ wgam,
               const bf16*  __restrict__ wprj,
               const float* __restrict__ biasB,
               float* __restrict__ out) {
  extern __shared__ char smem[];
  bf16*  xs  = (bf16*)(smem + XS_OFF);   // x staging; later attention output
  bf16*  qks = (bf16*)(smem + QK_OFF);   // q (cols 0..511, pre-scaled) | k (512..1023); later y
  bf16*  vts = (bf16*)(smem + VT_OFF);   // v^T [channel][token]
  float* ss  = (float*)(smem + SS_OFF);  // per-head scores 64x64
  bf16*  ps  = (bf16*)(smem + PS_OFF);   // per-head probs 64x64

  const int tid  = threadIdx.x;
  const int wave = tid >> 5;
  const int lane = tid & 31;
  const int li   = lane & 15;
  const int half = lane >> 4;
  const int b    = blockIdx.x;

  const v8f vzero = {0.f, 0.f, 0.f, 0.f, 0.f, 0.f, 0.f, 0.f};

  // ---- Phase A: stage x window fp32 -> bf16, zero pad rows 49..63 ----
  {
    const float4* xf = (const float4*)(x + (size_t)b * WIN_N * CDIM);
    for (int i = tid; i < WIN_N * CDIM / 4; i += 256) {
      float4 v = xf[i];
      int row = (i * 4) / CDIM, col = (i * 4) % CDIM;
      v4bf o = {(bf16)v.x, (bf16)v.y, (bf16)v.z, (bf16)v.w};
      *(v4bf*)(xs + row * XS_LD + col) = o;
    }
    for (int i = tid; i < (NPAD - WIN_N) * XS_LD; i += 256)
      xs[WIN_N * XS_LD + i] = (bf16)0.0f;
  }
  __syncthreads();

  // ---- Phase B: QKV GEMM (64x512 @ 512x1536) ----
  for (int t = 0; t < 12; ++t) {
    int nt = wave * 12 + t;
    int nb = nt * 16;                       // output channel base 0..1535
    v8f acc[4] = {vzero, vzero, vzero, vzero};
    gemm_nt64(xs, XS_LD, wqkv + (size_t)nb * CDIM, li, half, acc);
    int col = nb + li;
    float bias = b_qkv[col];
    if (nb < 2 * CDIM) {                    // q or k -> row-major qks
      bool isq = nb < CDIM;
#pragma unroll
      for (int m = 0; m < 4; ++m)
#pragma unroll
        for (int r = 0; r < 8; ++r) {
          float v = acc[m][r] + bias;
          if (isq) v *= 0.25f;              // scale = num_heads^-0.5
          qks[(m * 16 + half * 8 + r) * QK_LD + col] = (bf16)v;
        }
    } else {                                // v -> transposed vts (vector store)
#pragma unroll
      for (int m = 0; m < 4; ++m) {
        v8bf o;
#pragma unroll
        for (int r = 0; r < 8; ++r) o[r] = (bf16)(acc[m][r] + bias);
        *(v8bf*)(vts + (col - 2 * CDIM) * VT_LD + m * 16 + half * 8) = o;
      }
    }
  }
  __syncthreads();

  // ---- Phase C: attention per head ----
  for (int h = 0; h < HEADS; ++h) {
    // scores: 16 tiles (4i x 4j), 2 per wave; K = 32 = hd (single WMMA)
#pragma unroll
    for (int t = 0; t < 2; ++t) {
      int tt = wave * 2 + t;
      int it = tt >> 2, jt = tt & 3;
      const bf16* ap = qks + (it * 16 + li) * QK_LD + h * HD + half * 16;
      const bf16* bp = qks + (jt * 16 + li) * QK_LD + CDIM + h * HD + half * 16;
      v8f sc = wmma_bf16(frag_a(ap), frag_b(bp), vzero);
      const float* bb = biasB + h * (NPAD * NPAD) + jt * 16 + li;
#pragma unroll
      for (int r = 0; r < 8; ++r) {
        int i = it * 16 + half * 8 + r;
        ss[i * SS_LD + jt * 16 + li] = sc[r] + bb[i * NPAD];
      }
    }
    __syncthreads();

    // softmax over j<49, one row per thread (rows >=49 are junk but finite)
    if (tid < NPAD) {
      int row = tid;
      float mx = -1e30f;
      for (int j = 0; j < WIN_N; ++j) mx = fmaxf(mx, ss[row * SS_LD + j]);
      float sum = 0.f;
      for (int j = 0; j < WIN_N; ++j) sum += __expf(ss[row * SS_LD + j] - mx);
      float inv = 1.0f / sum;
      for (int j = 0; j < WIN_N; ++j)
        ps[row * PS_LD + j] = (bf16)(__expf(ss[row * SS_LD + j] - mx) * inv);
      for (int j = WIN_N; j < NPAD; ++j) ps[row * PS_LD + j] = (bf16)0.0f;
    }
    __syncthreads();

    // attn @ v : 8 tiles (4i x 2n), one per wave; K = 64 (2 steps)
    {
      int it = wave & 3, nt = wave >> 2;
      v8f acc = vzero;
#pragma unroll
      for (int ks = 0; ks < 2; ++ks) {
        const bf16* ap = ps + (it * 16 + li) * PS_LD + ks * 32 + half * 16;
        const bf16* bp = vts + (h * HD + nt * 16 + li) * VT_LD + ks * 32 + half * 16;
        acc = wmma_bf16(frag_a(ap), frag_b(bp), acc);
      }
      int col = h * HD + nt * 16 + li;
#pragma unroll
      for (int r = 0; r < 8; ++r)
        xs[(it * 16 + half * 8 + r) * XS_LD + col] = (bf16)acc[r];
    }
    __syncthreads();
  }

  // ---- Phase D: gamma GEMM  (attn-out in xs -> y in qks region) ----
  bf16* ys = qks;  // reuse q|k region, stride XS_LD
  for (int t = 0; t < 4; ++t) {
    int nb = (wave * 4 + t) * 16;
    v8f acc[4] = {vzero, vzero, vzero, vzero};
    gemm_nt64(xs, XS_LD, wgam + (size_t)nb * CDIM, li, half, acc);
    float bias = b_gam[nb + li];
#pragma unroll
    for (int m = 0; m < 4; ++m)
#pragma unroll
      for (int r = 0; r < 8; ++r)
        ys[(m * 16 + half * 8 + r) * XS_LD + nb + li] = (bf16)(acc[m][r] + bias);
  }
  __syncthreads();

  // ---- Phase E: proj GEMM -> global fp32 output (rows < 49 only) ----
  float* orow = out + (size_t)b * WIN_N * CDIM;
  for (int t = 0; t < 4; ++t) {
    int nb = (wave * 4 + t) * 16;
    v8f acc[4] = {vzero, vzero, vzero, vzero};
    gemm_nt64(ys, XS_LD, wprj + (size_t)nb * CDIM, li, half, acc);
    float bias = b_prj[nb + li];
#pragma unroll
    for (int m = 0; m < 4; ++m)
#pragma unroll
      for (int r = 0; r < 8; ++r) {
        int row = m * 16 + half * 8 + r;
        if (row < WIN_N) orow[row * CDIM + nb + li] = acc[m][r] + bias;
      }
  }
}

// ---------------------------------------------------------------------------
extern "C" void kernel_launch(void* const* d_in, const int* in_sizes, int n_in,
                              void* d_out, int out_size, void* d_ws, size_t ws_size,
                              hipStream_t stream) {
  (void)in_sizes; (void)n_in; (void)out_size; (void)ws_size;
  const float* x    = (const float*)d_in[0];
  const float* wqkv = (const float*)d_in[1];
  const float* bqkv = (const float*)d_in[2];
  const float* wgam = (const float*)d_in[3];
  const float* bgam = (const float*)d_in[4];
  const float* wprj = (const float*)d_in[5];
  const float* bprj = (const float*)d_in[6];
  const float* btab = (const float*)d_in[7];
  const int*   ridx = (const int*)d_in[8];

  char* ws = (char*)d_ws;
  bf16*  wqkv_b = (bf16*)ws;                                    // 1536*512 bf16
  bf16*  wgam_b = (bf16*)(ws + 786432 * 2);                     // 512*512 bf16
  bf16*  wprj_b = (bf16*)(ws + 786432 * 2 + 262144 * 2);        // 512*512 bf16
  float* biasB  = (float*)(ws + 786432 * 2 + 262144 * 4);       // 16*64*64 f32

  prep_kernel<<<512, 256, 0, stream>>>(wqkv, wgam, wprj, btab, ridx,
                                       wqkv_b, wgam_b, wprj_b, biasB);

  hipFuncSetAttribute((const void*)win_attn_fused,
                      hipFuncAttributeMaxDynamicSharedMemorySize, SMEM_BYTES);
  win_attn_fused<<<NWIN, 256, SMEM_BYTES, stream>>>(
      x, bqkv, bgam, bprj, wqkv_b, wgam_b, wprj_b, biasB, (float*)d_out);
}
